// ModeT_cu_30494267802028
// MI455X (gfx1250) — matrix operand order repaired and checked by reading, not compiled, not run on hardware
//
#include <hip/hip_runtime.h>
#include <hip/hip_bf16.h>
#include <math.h>
#include <stdint.h>

// ---------------------------------------------------------------------------
// Types for CDNA5 WMMA (gfx1250, wave32)
// ---------------------------------------------------------------------------
typedef __bf16 bf16x16 __attribute__((ext_vector_type(16)));
typedef float  f32x8   __attribute__((ext_vector_type(8)));

#define LEAK 0.1f
#define EPSF 1e-5f

__device__ __forceinline__ float leaky_op(float v) { return v >= 0.f ? v : LEAK * v; }

// ---------------------------------------------------------------------------
// Conv3d 3x3x3 SAME as implicit GEMM, bf16 WMMA 16x16x32, f32 accumulate.
//
// K padded PER CHANNEL to 32 (27 taps + 5 zeros): every 32-wide WMMA K-chunk
// is exactly one input channel, so the tap id of each fragment element is
// lane-constant for the whole kernel. Per lane we precompute:
//   adr[e] : CLAMPED in-bounds gather address (invalid -> own voxel)
//   msk[e] : 0/1 float multiplier for zero-padding
// so the inner loop issues 32 UNCONDITIONAL global_load_b32 (one wait for the
// whole batch) followed by branchless v_mul + cvt_pk_bf16 packing — no
// per-element EXEC-mask branches / serialized waits.
// Weights are staged into LDS pre-swizzled in B-fragment layout so each lane
// loads its fragment as one 32B vector (2x ds_load_b128).
// Each wave owns an M=32 x N=16 tile: two A fragments / accumulators share
// one B fragment (2 back-to-back v_wmma per chunk).
// ---------------------------------------------------------------------------
#define CONV_WAVES 4
#define CCH 8      // input channels staged per LDS round (8 * 512 bf16 = 8 KB)

__global__ void __launch_bounds__(32 * CONV_WAVES)
conv3d_wmma_k(const float* __restrict__ x, const float* __restrict__ wg,
              const float* __restrict__ bias, float* __restrict__ y,
              int Cin, int Cout, int D, int H, int W)
{
  // B-fragment-layout weight stage: [CCH][32 lanes][16 elems]
  __shared__ __align__(16) __bf16 wlds[CCH * 32 * 16];

  const int n0   = blockIdx.y * 16;          // output-channel tile base
  const int tid  = threadIdx.x;
  const int wave = tid >> 5;
  const int lane = tid & 31;

  const int HW   = H * W;
  const int nvox = D * HW;
  const int m0   = (blockIdx.x * CONV_WAVES + wave) * 32;   // voxel tile base

  // Two A-fragment rows owned by this lane (tiles m0..m0+15 and m0+16..m0+31)
  const int  mA = m0 + (lane & 15);
  const int  mB = m0 + 16 + (lane & 15);
  const bool mvA = (mA < nvox);
  const bool mvB = (mB < nvox);
  const int  posA = mvA ? mA : 0;
  const int  posB = mvB ? mB : 0;
  const int  vzA = posA / HW, vyA = (posA / W) % H, vxA = posA % W;
  const int  vzB = posB / HW, vyB = (posB / W) % H, vxB = posB % W;

  const int khalfA = (lane < 16) ? 0 : 8;    // A: K {0..7,16..23} vs {8..15,24..31}
  const int nlane  = lane & 15;              // B/D: N = lane%16

  // ---- per-lane tap tables, hoisted out of the K loop -------------------
  // adr*: clamped (always-safe) gather index; msk*: 0/1 zero-pad multiplier.
  int   adrA[16], adrB[16];
  float mskA[16], mskB[16];
#pragma unroll
  for (int e = 0; e < 16; ++e) {
    const int toff = ((e < 8) ? e : e + 8) + khalfA;   // tap id 0..31
    adrA[e] = posA; adrB[e] = posB;
    mskA[e] = 0.f;  mskB[e] = 0.f;
    if (toff < 27) {
      const int dz = toff / 9 - 1;
      const int dy = (toff / 3) % 3 - 1;
      const int dx = toff % 3 - 1;
      const int off = dz * HW + dy * W + dx;
      if (mvA && (unsigned)(vzA + dz) < (unsigned)D &&
          (unsigned)(vyA + dy) < (unsigned)H &&
          (unsigned)(vxA + dx) < (unsigned)W) {
        adrA[e] = posA + off;
        mskA[e] = 1.f;
      }
      if (mvB && (unsigned)(vzB + dz) < (unsigned)D &&
          (unsigned)(vyB + dy) < (unsigned)H &&
          (unsigned)(vxB + dx) < (unsigned)W) {
        adrB[e] = posB + off;
        mskB[e] = 1.f;
      }
    }
  }

  f32x8 acc0 = {};
  f32x8 acc1 = {};

  for (int cin0 = 0; cin0 < Cin; cin0 += CCH) {
    const int nch = (cin0 + CCH < Cin) ? CCH : (Cin - cin0);

    __syncthreads();
    // ---- stage weights pre-swizzled into B-fragment layout ----
    // slot(c_local, lane_s, e) holds w[n0 + (lane_s&15)][cin0+c_local][tap]
    // with tap = ((e<8)?e:e+8) + ((lane_s<16)?0:8)  (zeros for tap>=27).
    for (int idx = tid; idx < nch * 512; idx += 32 * CONV_WAVES) {
      const int c_local = idx >> 9;
      const int rem     = idx & 511;
      const int lane_s  = rem >> 4;
      const int e       = rem & 15;
      const int tap     = ((e < 8) ? e : e + 8) + ((lane_s < 16) ? 0 : 8);
      const int n       = lane_s & 15;
      float v = 0.f;
      if (tap < 27 && (n0 + n) < Cout)
        v = wg[((size_t)(n0 + n) * Cin + (cin0 + c_local)) * 27 + tap];
      wlds[idx] = (__bf16)v;
    }
    __syncthreads();

    // prefetch next channel-group weights (global_prefetch_b8)
    if (cin0 + CCH < Cin && tid < 32 && n0 < Cout) {
      __builtin_prefetch(&wg[((size_t)n0 * Cin + (cin0 + CCH)) * 27 + tid * 4], 0, 0);
    }

    for (int c = 0; c < nch; ++c) {
      const float* __restrict__ xa = x + (size_t)(cin0 + c) * nvox;
      // ---- batched, UNCONDITIONAL neighbor gathers (one wait per batch) --
      float fA[16], fB[16];
#pragma unroll
      for (int e = 0; e < 16; ++e) fA[e] = xa[adrA[e]];
#pragma unroll
      for (int e = 0; e < 16; ++e) fB[e] = xa[adrB[e]];
      // ---- branchless zero-pad + pack to bf16 ----
      bf16x16 aA, aB;
#pragma unroll
      for (int e = 0; e < 16; ++e) {
        aA[e] = (__bf16)(fA[e] * mskA[e]);
        aB[e] = (__bf16)(fB[e] * mskB[e]);
      }
      // ---- B fragment: one 32B vector load from LDS ----
      const bf16x16 bfrag =
          *(const bf16x16*)&wlds[((size_t)c * 32 + lane) * 16];

      acc0 = __builtin_amdgcn_wmma_f32_16x16x32_bf16(false, aA, false, bfrag,
                                                     (short)0, acc0, false, false);
      acc1 = __builtin_amdgcn_wmma_f32_16x16x32_bf16(false, aB, false, bfrag,
                                                     (short)0, acc1, false, false);
    }
  }

  // ---- write D tiles: element r -> M = r + 8*(lane>=16), N = lane%16 ----
  const int n = n0 + nlane;
  if (n < Cout) {
    const float bv = bias[n];
    const int mbase0 = m0 + ((lane < 16) ? 0 : 8);
    const int mbase1 = mbase0 + 16;
#pragma unroll
    for (int r = 0; r < 8; ++r) {
      const int mo0 = mbase0 + r;
      const int mo1 = mbase1 + r;
      if (mo0 < nvox) y[(size_t)n * nvox + mo0] = acc0[r] + bv;
      if (mo1 < nvox) y[(size_t)n * nvox + mo1] = acc1[r] + bv;
    }
  }
}

// ---------------------------------------------------------------------------
// Fused instance-norm (+leaky) : one block per channel, two-pass, in place
// ---------------------------------------------------------------------------
__global__ void inorm_leaky_k(float* __restrict__ buf, int N)
{
  const int c = blockIdx.x;
  float* p = buf + (size_t)c * N;
  __shared__ float rs[32], rs2[32];

  float s = 0.f, s2 = 0.f;
  for (int i = threadIdx.x; i < N; i += blockDim.x) { float v = p[i]; s += v; s2 += v * v; }
#pragma unroll
  for (int o = 16; o > 0; o >>= 1) { s += __shfl_xor(s, o, 32); s2 += __shfl_xor(s2, o, 32); }
  const int wv = threadIdx.x >> 5, ln = threadIdx.x & 31;
  if (ln == 0) { rs[wv] = s; rs2[wv] = s2; }
  __syncthreads();
  const int nw = blockDim.x >> 5;
  if (wv == 0) {
    s  = (ln < nw) ? rs[ln]  : 0.f;
    s2 = (ln < nw) ? rs2[ln] : 0.f;
#pragma unroll
    for (int o = 16; o > 0; o >>= 1) { s += __shfl_xor(s, o, 32); s2 += __shfl_xor(s2, o, 32); }
    if (ln == 0) { rs[0] = s; rs2[0] = s2; }
  }
  __syncthreads();
  const float mean = rs[0] / (float)N;
  const float var  = rs2[0] / (float)N - mean * mean;
  const float inv  = rsqrtf(var + EPSF);
  for (int i = threadIdx.x; i < N; i += blockDim.x)
    p[i] = leaky_op((p[i] - mean) * inv);
}

__global__ void leaky_k(float* __restrict__ b, long n)
{
  long i = (long)blockIdx.x * blockDim.x + threadIdx.x;
  if (i < n) b[i] = leaky_op(b[i]);
}

__global__ void add_k(const float* __restrict__ a, const float* __restrict__ b,
                      float* __restrict__ o, long n)
{
  long i = (long)blockIdx.x * blockDim.x + threadIdx.x;
  if (i < n) o[i] = a[i] + b[i];
}

// ---------------------------------------------------------------------------
// 2x2x2 average pool
// ---------------------------------------------------------------------------
__global__ void avgpool2_k(const float* __restrict__ x, float* __restrict__ y,
                           int C, int D, int H, int W)
{
  const int Do = D / 2, Ho = H / 2, Wo = W / 2;
  const long No = (long)Do * Ho * Wo;
  long idx = (long)blockIdx.x * blockDim.x + threadIdx.x;
  if (idx >= (long)C * No) return;
  const int c = (int)(idx / No);
  long r = idx - (long)c * No;
  const int oz = (int)(r / ((long)Ho * Wo));
  const int oy = (int)((r / Wo) % Ho);
  const int ox = (int)(r % Wo);
  const float* xc = x + (size_t)c * D * H * W;
  float s = 0.f;
#pragma unroll
  for (int dz = 0; dz < 2; ++dz)
#pragma unroll
    for (int dy = 0; dy < 2; ++dy)
#pragma unroll
      for (int dx = 0; dx < 2; ++dx)
        s += xc[((size_t)(2 * oz + dz) * H + (2 * oy + dy)) * W + (2 * ox + dx)];
  y[idx] = s * 0.125f;
}

// ---------------------------------------------------------------------------
// up2 : trilinear resize to 2x on every axis, pos = i*(n-1)/(2n-1)  (+scale)
// ---------------------------------------------------------------------------
__global__ void up2_k(const float* __restrict__ x, float* __restrict__ y,
                      int C, int D, int H, int W, float scale)
{
  const int Do = 2 * D, Ho = 2 * H, Wo = 2 * W;
  const long No = (long)Do * Ho * Wo;
  long idx = (long)blockIdx.x * blockDim.x + threadIdx.x;
  if (idx >= (long)C * No) return;
  const int c = (int)(idx / No);
  long r = idx - (long)c * No;
  const int oz = (int)(r / ((long)Ho * Wo));
  const int oy = (int)((r / Wo) % Ho);
  const int ox = (int)(r % Wo);
  const float pz = oz * ((float)(D - 1) / (float)(Do - 1));
  const float py = oy * ((float)(H - 1) / (float)(Ho - 1));
  const float px = ox * ((float)(W - 1) / (float)(Wo - 1));
  int iz = (int)floorf(pz); iz = iz < 0 ? 0 : (iz > D - 2 ? D - 2 : iz);
  int iy = (int)floorf(py); iy = iy < 0 ? 0 : (iy > H - 2 ? H - 2 : iy);
  int ix = (int)floorf(px); ix = ix < 0 ? 0 : (ix > W - 2 ? W - 2 : ix);
  const float tz = pz - iz, ty = py - iy, tx = px - ix;
  const float* xc = x + (size_t)c * D * H * W;
#define XV(a, b, cc) xc[((size_t)(a) * H + (b)) * W + (cc)]
  const float c00 = XV(iz,     iy,     ix) * (1.f - tx) + XV(iz,     iy,     ix + 1) * tx;
  const float c01 = XV(iz,     iy + 1, ix) * (1.f - tx) + XV(iz,     iy + 1, ix + 1) * tx;
  const float c10 = XV(iz + 1, iy,     ix) * (1.f - tx) + XV(iz + 1, iy,     ix + 1) * tx;
  const float c11 = XV(iz + 1, iy + 1, ix) * (1.f - tx) + XV(iz + 1, iy + 1, ix + 1) * tx;
#undef XV
  const float c0 = c00 * (1.f - ty) + c01 * ty;
  const float c1 = c10 * (1.f - ty) + c11 * ty;
  y[idx] = scale * (c0 * (1.f - tz) + c1 * tz);
}

// ---------------------------------------------------------------------------
// warp : out[c,v] = trilinear sample of src at (grid + flow), zero outside
// ---------------------------------------------------------------------------
__global__ void warp_k(const float* __restrict__ src, const float* __restrict__ flow,
                       float* __restrict__ out, int C, int D, int H, int W)
{
  const long Nv = (long)D * H * W;
  long v = (long)blockIdx.x * blockDim.x + threadIdx.x;
  if (v >= Nv) return;
  const int z  = (int)(v / ((long)H * W));
  const int yy = (int)((v / W) % H);
  const int xx = (int)(v % W);
  const float d = z  + flow[v];
  const float h = yy + flow[Nv + v];
  const float w = xx + flow[2 * Nv + v];
  const float d0 = floorf(d), h0 = floorf(h), w0 = floorf(w);
  float wt8[8];
  long id8[8];
  int q = 0;
#pragma unroll
  for (int dd = 0; dd < 2; ++dd)
#pragma unroll
    for (int hh = 0; hh < 2; ++hh)
#pragma unroll
      for (int ww = 0; ww < 2; ++ww) {
        const float fd = d0 + dd, fh = h0 + hh, fw = w0 + ww;
        float wt = (1.f - fabsf(d - fd)) * (1.f - fabsf(h - fh)) * (1.f - fabsf(w - fw));
        const bool valid = (fd >= 0.f) && (fd < (float)D) && (fh >= 0.f) &&
                           (fh < (float)H) && (fw >= 0.f) && (fw < (float)W);
        if (!valid) wt = 0.f;
        int gi = (int)fd; gi = gi < 0 ? 0 : (gi > D - 1 ? D - 1 : gi);
        int gj = (int)fh; gj = gj < 0 ? 0 : (gj > H - 1 ? H - 1 : gj);
        int gk = (int)fw; gk = gk < 0 ? 0 : (gk > W - 1 ? W - 1 : gk);
        wt8[q] = wt;
        id8[q] = ((long)gi * H + gj) * W + gk;
        ++q;
      }
  for (int ch = 0; ch < C; ++ch) {
    const float* sc = src + (size_t)ch * Nv;
    float a = 0.f;
#pragma unroll
    for (int t = 0; t < 8; ++t) a += sc[id8[t]] * wt8[t];
    out[(size_t)ch * Nv + v] = a;
  }
}

// ---------------------------------------------------------------------------
// proj : per-voxel linear (C->Co) + layer-norm over Co + affine, out voxel-major
// ---------------------------------------------------------------------------
__global__ void proj_k(const float* __restrict__ feat, const float* __restrict__ Wm,
                       const float* __restrict__ bb, const float* __restrict__ gg,
                       const float* __restrict__ be, float* __restrict__ out,
                       int C, int Co, int Nv)
{
  const int v = blockIdx.x * blockDim.x + threadIdx.x;
  if (v >= Nv) return;
  float yv[48];
  float mean = 0.f;
  for (int o = 0; o < Co; ++o) {
    float a = bb[o];
    const float* wr = Wm + (size_t)o * C;
    for (int c = 0; c < C; ++c) a += feat[(size_t)c * Nv + v] * wr[c];
    yv[o] = a;
    mean += a;
  }
  mean /= (float)Co;
  float var = 0.f;
  for (int o = 0; o < Co; ++o) { float dd = yv[o] - mean; var += dd * dd; }
  var /= (float)Co;
  const float inv = rsqrtf(var + EPSF);
  float* op = out + (size_t)v * Co;
  for (int o = 0; o < Co; ++o) op[o] = (yv[o] - mean) * inv * gg[o] + be[o];
}

// ---------------------------------------------------------------------------
// mode_attn : 27-neighbor attention, head dim = 6, V generated arithmetically
// q,k voxel-major (Nv, nh*6); out channel-major (nh*3, Nv)
// ---------------------------------------------------------------------------
__global__ void mode_attn_k(const float* __restrict__ q, const float* __restrict__ k,
                            const float* __restrict__ rpb, float* __restrict__ out,
                            int nh, int D, int H, int W)
{
  const long Nv = (long)D * H * W;
  long idx = (long)blockIdx.x * blockDim.x + threadIdx.x;
  if (idx >= Nv * nh) return;
  const int  head = (int)(idx / Nv);
  const long v    = idx - (long)head * Nv;
  const int  z  = (int)(v / ((long)H * W));
  const int  yy = (int)((v / W) % H);
  const int  xx = (int)(v % W);
  const int  C  = nh * 6;
  float qv[6];
#pragma unroll
  for (int e = 0; e < 6; ++e) qv[e] = q[v * C + head * 6 + e];

  float lg[27];
  float mx = -1e30f;
#pragma unroll
  for (int m = 0; m < 27; ++m) {
    const int dz = m / 9 - 1, dy = (m / 3) % 3 - 1, dx = m % 3 - 1;
    const int az = z + dz, ay = yy + dy, ax = xx + dx;
    float dot = 0.f;
    if ((unsigned)az < (unsigned)D && (unsigned)ay < (unsigned)H &&
        (unsigned)ax < (unsigned)W) {
      const long u = ((long)az * H + ay) * W + ax;
#pragma unroll
      for (int e = 0; e < 6; ++e) dot += qv[e] * k[u * C + head * 6 + e];
    }
    const float l = dot + rpb[head * 27 + m];
    lg[m] = l;
    mx = fmaxf(mx, l);
  }
  float s = 0.f;
#pragma unroll
  for (int m = 0; m < 27; ++m) { lg[m] = expf(lg[m] - mx); s += lg[m]; }
  const float inv = 1.f / s;
  float a0 = 0.f, a1 = 0.f, a2 = 0.f;
#pragma unroll
  for (int m = 0; m < 27; ++m) {
    const float p = lg[m] * inv;
    a0 += p * (float)(m / 9 - 1);
    a1 += p * (float)((m / 3) % 3 - 1);
    a2 += p * (float)(m % 3 - 1);
  }
  out[((size_t)(head * 3 + 0)) * Nv + v] = a0;
  out[((size_t)(head * 3 + 1)) * Nv + v] = a1;
  out[((size_t)(head * 3 + 2)) * Nv + v] = a2;
}

// ---------------------------------------------------------------------------
// cwm combine : softmax over nf weight maps, blend 3-vector groups, x2
// ---------------------------------------------------------------------------
__global__ void cwm_combine_k(const float* __restrict__ x, const float* __restrict__ lg,
                              float* __restrict__ out, int nf, long Nv)
{
  long v = (long)blockIdx.x * blockDim.x + threadIdx.x;
  if (v >= Nv) return;
  float p[8];
  float mx = -1e30f;
  for (int f = 0; f < nf; ++f) { p[f] = lg[(size_t)f * Nv + v]; mx = fmaxf(mx, p[f]); }
  float s = 0.f;
  for (int f = 0; f < nf; ++f) { p[f] = expf(p[f] - mx); s += p[f]; }
  const float inv = 1.f / s;
  for (int a = 0; a < 3; ++a) {
    float acc = 0.f;
    for (int f = 0; f < nf; ++f) acc += x[((size_t)(f * 3 + a)) * Nv + v] * p[f];
    out[(size_t)a * Nv + v] = 2.f * acc * inv;
  }
}

// ---------------------------------------------------------------------------
// Host orchestration
// ---------------------------------------------------------------------------
static inline unsigned cdivu(long a, long b) { return (unsigned)((a + b - 1) / b); }

extern "C" void kernel_launch(void* const* d_in, const int* in_sizes, int n_in,
                              void* d_out, int out_size, void* d_ws, size_t ws_size,
                              hipStream_t stream)
{
  (void)in_sizes; (void)n_in; (void)out_size; (void)ws_size;
#define PF(i) ((const float*)d_in[(i)])
  const float* moving = PF(0);
  const float* fixedv = PF(1);

  // dims per pyramid level
  const int D0 = 96, H0 = 112, W0 = 96;
  const int D1 = 48, H1 = 56,  W1 = 48;
  const int D2 = 24, H2 = 28,  W2 = 24;
  const int D3 = 12, H3 = 14,  W3 = 12;
  const int D4 = 6,  H4 = 7,   W4 = 6;
  const long Nv0 = (long)D0 * H0 * W0;
  const long Nv1 = (long)D1 * H1 * W1;
  const long Nv2 = (long)D2 * H2 * W2;
  const long Nv3 = (long)D3 * H3 * W3;
  const long Nv4 = (long)D4 * H4 * W4;

  // bump allocator over workspace
  char* wp = (char*)d_ws;
  auto alloc = [&](long nfl) -> float* {
    float* r = (float*)wp;
    wp += (((size_t)nfl * 4 + 255) / 256) * 256;
    return r;
  };

  float* M1 = alloc(8 * Nv0);   float* F1 = alloc(8 * Nv0);
  float* M2 = alloc(16 * Nv1);  float* F2 = alloc(16 * Nv1);
  float* M3 = alloc(32 * Nv2);  float* F3 = alloc(32 * Nv2);
  float* M4 = alloc(64 * Nv3);  float* F4 = alloc(64 * Nv3);
  float* M5 = alloc(128 * Nv4); float* F5 = alloc(128 * Nv4);
  float* SA  = alloc(8 * Nv0);  float* SB  = alloc(8 * Nv0);   // encoder ping-pong
  float* QB  = alloc(6 * Nv0);  float* KB  = alloc(6 * Nv0);   // proj outputs
  float* WA  = alloc(3 * Nv0);                                  // attention out
  float* XUP = alloc(Nv0);                                      // cwm upsample
  float* HA  = alloc(2 * Nv0);  float* HB  = alloc(2 * Nv0);    // cwm conv bufs
  float* FL1 = alloc(3 * Nv0);  float* FL2 = alloc(3 * Nv0);
  float* FL3 = alloc(3 * Nv0);
  float* WRP = alloc(8 * Nv0);                                  // warped features

  // ---- launch helpers ----
  auto conv = [&](const float* x, int wi, float* y, int Ci, int Co, int d, int h, int w) {
    const long nv = (long)d * h * w;
    dim3 g(cdivu(nv, 32 * CONV_WAVES), (unsigned)((Co + 15) / 16));
    conv3d_wmma_k<<<g, 32 * CONV_WAVES, 0, stream>>>(x, PF(wi), PF(wi + 1), y, Ci, Co, d, h, w);
  };
  auto inorm = [&](float* b, int C, long nv) {
    inorm_leaky_k<<<(unsigned)C, 256, 0, stream>>>(b, (int)nv);
  };
  auto leaky = [&](float* b, long n) {
    leaky_k<<<cdivu(n, 256), 256, 0, stream>>>(b, n);
  };
  auto pool = [&](const float* x, float* y, int C, int d, int h, int w) {
    const long n = (long)C * (d / 2) * (h / 2) * (w / 2);
    avgpool2_k<<<cdivu(n, 256), 256, 0, stream>>>(x, y, C, d, h, w);
  };
  auto up2 = [&](const float* x, float* y, int C, int d, int h, int w, float sc) {
    const long n = (long)C * 8 * d * h * w;
    up2_k<<<cdivu(n, 256), 256, 0, stream>>>(x, y, C, d, h, w, sc);
  };
  auto warp = [&](const float* src, const float* fl, float* o, int C, int d, int h, int w) {
    const long nv = (long)d * h * w;
    warp_k<<<cdivu(nv, 256), 256, 0, stream>>>(src, fl, o, C, d, h, w);
  };
  auto add = [&](const float* a, const float* b, float* o, long n) {
    add_k<<<cdivu(n, 256), 256, 0, stream>>>(a, b, o, n);
  };
  auto proj = [&](const float* feat, int pi, float* o, int C, int Co, long nv) {
    proj_k<<<cdivu(nv, 256), 256, 0, stream>>>(feat, PF(pi), PF(pi + 1), PF(pi + 2),
                                               PF(pi + 3), o, C, Co, (int)nv);
  };
  auto attn = [&](const float* qb, const float* kb, int ri, float* o, int nh,
                  int d, int h, int w) {
    const long n = (long)d * h * w * nh;
    mode_attn_k<<<cdivu(n, 256), 256, 0, stream>>>(qb, kb, PF(ri), o, nh, d, h, w);
  };
  // cwm: up2 -> conv_ins a -> conv_ins b -> conv c -> softmax combine (x2)
  auto cwm = [&](const float* win, int Cw, int pa, int pb, int pc, int nf,
                 int d, int h, int w, float* o) {
    const int d2 = 2 * d, h2 = 2 * h, w2 = 2 * w;
    const long nvO = (long)d2 * h2 * w2;
    const int Ca = 2 * Cw;
    up2(win, XUP, Cw, d, h, w, 1.0f);
    conv(XUP, pa, HA, Cw, Ca, d2, h2, w2); inorm(HA, Ca, nvO);
    conv(HA, pb, HB, Ca, Ca, d2, h2, w2);  inorm(HB, Ca, nvO);
    conv(HB, pc, HA, Ca, nf, d2, h2, w2);
    cwm_combine_k<<<cdivu(nvO, 256), 256, 0, stream>>>(XUP, HA, o, nf, nvO);
  };

  // ---- encoder (shared weights, params e0a..e4b = indices 2..23) ----
  auto encode = [&](const float* in, float* X1, float* X2, float* X3, float* X4, float* X5) {
    conv(in, 2, SA, 1, 4, D0, H0, W0);   leaky(SA, 4 * Nv0);            // e0a (conv_blk)
    conv(SA, 4, SB, 4, 8, D0, H0, W0);   inorm(SB, 8, Nv0);             // e0b
    conv(SB, 6, X1, 8, 8, D0, H0, W0);   inorm(X1, 8, Nv0);             // e0c -> x0
    pool(X1, SA, 8, D0, H0, W0);
    conv(SA, 8,  SB, 8, 16, D1, H1, W1); inorm(SB, 16, Nv1);            // e1a
    conv(SB, 10, X2, 16, 16, D1, H1, W1); inorm(X2, 16, Nv1);           // e1b -> x1
    pool(X2, SA, 16, D1, H1, W1);
    conv(SA, 12, SB, 16, 32, D2, H2, W2); inorm(SB, 32, Nv2);           // e2a
    conv(SB, 14, X3, 32, 32, D2, H2, W2); inorm(X3, 32, Nv2);           // e2b -> x2
    pool(X3, SA, 32, D2, H2, W2);
    conv(SA, 16, SB, 32, 64, D3, H3, W3); inorm(SB, 64, Nv3);           // e3a
    conv(SB, 18, X4, 64, 64, D3, H3, W3); inorm(X4, 64, Nv3);           // e3b -> x3
    pool(X4, SA, 64, D3, H3, W3);
    conv(SA, 20, SB, 64, 128, D4, H4, W4);  inorm(SB, 128, Nv4);        // e4a
    conv(SB, 22, X5, 128, 128, D4, H4, W4); inorm(X5, 128, Nv4);        // e4b -> x4
  };

  encode(moving, M1, M2, M3, M4, M5);
  encode(fixedv, F1, F2, F3, F4, F5);

  // ---- level 5 : attn(F5,M5) -> cwm5 -> flow @ L3 ----
  proj(F5, 40, QB, 128, 48, Nv4);
  proj(M5, 40, KB, 128, 48, Nv4);
  attn(QB, KB, 48, WA, 8, D4, H4, W4);
  cwm(WA, 24, 49, 51, 53, 8, D4, H4, W4, FL1);          // FL1 @ L3

  // ---- level 4 ----
  warp(M4, FL1, WRP, 64, D3, H3, W3);
  proj(F4, 36, QB, 64, 24, Nv3);
  proj(WRP, 36, KB, 64, 24, Nv3);
  attn(QB, KB, 47, WA, 4, D3, H3, W3);
  cwm(WA, 12, 55, 57, 59, 4, D3, H3, W3, FL2);          // FL2 = w @ L2
  up2(FL1, FL3, 3, D3, H3, W3, 2.0f);                   // FL3 = up2(2*flow) @ L2
  warp(FL3, FL2, FL1, 3, D2, H2, W2);                   // FL1 = warp(up2(2f), w)
  add(FL1, FL2, FL1, 3 * Nv2);                          // flow @ L2

  // ---- level 3 ----
  warp(M3, FL1, WRP, 32, D2, H2, W2);
  proj(F3, 32, QB, 32, 12, Nv2);
  proj(WRP, 32, KB, 32, 12, Nv2);
  attn(QB, KB, 46, WA, 2, D2, H2, W2);
  cwm(WA, 6, 61, 63, 65, 2, D2, H2, W2, FL2);           // FL2 = w @ L1
  up2(FL1, FL3, 3, D2, H2, W2, 2.0f);                   // FL3 @ L1
  warp(FL3, FL2, FL1, 3, D1, H1, W1);
  add(FL1, FL2, FL1, 3 * Nv1);                          // flow @ L1

  // ---- level 2 (no cwm) ----
  warp(M2, FL1, WRP, 16, D1, H1, W1);
  proj(F2, 28, QB, 16, 6, Nv1);
  proj(WRP, 28, KB, 16, 6, Nv1);
  attn(QB, KB, 45, WA, 1, D1, H1, W1);                  // w @ L1 (3ch)
  warp(FL1, WA, FL3, 3, D1, H1, W1);                    // warp(flow, w)
  add(FL3, WA, FL2, 3 * Nv1);
  up2(FL2, FL1, 3, D1, H1, W1, 2.0f);                   // flow @ L0

  // ---- level 1 ----
  warp(M1, FL1, WRP, 8, D0, H0, W0);
  proj(F1, 24, QB, 8, 6, Nv0);
  proj(WRP, 24, KB, 8, 6, Nv0);
  attn(QB, KB, 44, WA, 1, D0, H0, W0);                  // w @ L0 (3ch)
  float* outw  = (float*)d_out;                         // warped moving (Nv0)
  float* outfl = outw + Nv0;                            // final flow (3*Nv0)
  warp(FL1, WA, FL2, 3, D0, H0, W0);
  add(FL2, WA, outfl, 3 * Nv0);                         // final flow -> d_out
  warp(moving, outfl, outw, 1, D0, H0, W0);             // warped moving -> d_out
#undef PF
}
